// TreeLSTM_81338090651819
// MI455X (gfx1250) — compile-verified, matrix operand order checked by reading, not run on hardware
//
#include <hip/hip_runtime.h>
#include <hip/hip_bf16.h>
#include <math.h>

#define Bsz 256
#define Tn  128
#define En  300
#define Hn  512
#define KP  320            // E padded to multiple of 32
#define NR  (5*Hn)         // 2560
#define KR  (2*Hn)         // 1024
#define MT  (Bsz*Tn)       // 32768
#define SMAX 257           // max steps supported (steps = 2T-1 = 255)
#define ZSLOT (Tn + SMAX - 1)   // 384: dedicated all-zeros slot
#define SLOTS (Tn + SMAX)       // 385 slots per batch row

typedef __attribute__((ext_vector_type(16))) __bf16 v16bf;
typedef __attribute__((ext_vector_type(8)))  float  v8f;

union Frag { v16bf v; unsigned int w[8]; };

__device__ __forceinline__ unsigned short f2bf(float f) {
  union { float f; unsigned u; } x; x.f = f;
  unsigned r = x.u + 0x7FFFu + ((x.u >> 16) & 1u);   // round-to-nearest-even
  return (unsigned short)(r >> 16);
}
__device__ __forceinline__ float sigmoidf_(float x) { return 1.0f / (1.0f + __expf(-x)); }
__device__ __forceinline__ int   imax_(int a, int b) { return a > b ? a : b; }
__device__ __forceinline__ int   imin_(int a, int b) { return a < b ? a : b; }

// K offset (in bf16 elements) within a 32-wide chunk for fragment pair p, per
// ISA 16-bit A/B layout: lanes 0-15 hold K {0..7,16..23}, lanes 16-31 {8..15,24..31}.
#define KOFF(p, hi) ((((p) < 4) ? 0 : 16) + (hi) * 8 + (((p) & 3) * 2))

// ---------------- schedule precompute: stack of slot ids ----------------
// Reduce at step t writes fresh slot Tn+t; leaves are slots 0..Tn-1; ZSLOT = zeros.
__global__ void k_sched(const int* __restrict__ trans, int steps,
                        int* __restrict__ lslot, int* __restrict__ rslot,
                        int* __restrict__ top, int* __restrict__ flags) {
  int b = threadIdx.x;
  if (b < 256) flags[b] = 0;
  __syncthreads();
  if (b >= Bsz) return;
  int stk[Tn];
  for (int i = 0; i < Tn; ++i) stk[i] = ZSLOT;
  int sp = 0, bp = Tn;
  for (int t = 0; t < steps; ++t) {
    int l = stk[imax_(sp - 2, 0)];
    int r = stk[imax_(sp - 1, 0)];
    lslot[t * Bsz + b] = l;
    rslot[t * Bsz + b] = r;
    int tr = trans[(size_t)t * Bsz + b];
    if (tr == 0) {
      int pos = imax_(imin_(sp, Tn - 1), 0);
      stk[pos] = bp - 1;                 // push leaf slot
      sp += 1; bp -= 1;
    } else {
      int pos = imax_(sp - 2, 0);
      stk[pos] = Tn + t;                 // push reduce-output slot
      sp -= 1;
      atomicOr(&flags[t], 1);
    }
  }
  top[b] = stk[imax_(sp - 1, 0)];
}

// ---------------- conversion / init kernels ----------------

__global__ void k_conv_x(const float* __restrict__ x, unsigned short* __restrict__ xb) {
  size_t idx = (size_t)blockIdx.x * blockDim.x + threadIdx.x;
  if (idx >= (size_t)MT * KP) return;
  size_t m = idx / KP, k = idx % KP;
  xb[idx] = (k < En) ? f2bf(x[m * En + k]) : (unsigned short)0;
}

__global__ void k_conv_wpg(const float* __restrict__ Wp, const float* __restrict__ Wg,
                           unsigned short* __restrict__ Wpt, unsigned short* __restrict__ Wgt) {
  size_t idx = (size_t)blockIdx.x * blockDim.x + threadIdx.x;
  if (idx >= (size_t)Hn * KP) return;
  size_t n = idx / KP, k = idx % KP;
  Wpt[idx] = (k < En) ? f2bf(Wp[k * Hn + n]) : (unsigned short)0;
  Wgt[idx] = (k < En) ? f2bf(Wg[k * Hn + n]) : (unsigned short)0;
}

__global__ void k_conv_wr(const float* __restrict__ Wr, unsigned short* __restrict__ Wrt) {
  size_t idx = (size_t)blockIdx.x * blockDim.x + threadIdx.x;
  if (idx >= (size_t)NR * KR) return;
  size_t n = idx / KR, k = idx % KR;
  Wrt[idx] = f2bf(Wr[k * NR + n]);       // transposed: row n holds K contiguously
}

__global__ void k_zero_z(float* __restrict__ cf, float* __restrict__ hf,
                         unsigned short* __restrict__ hb) {
  int idx = blockIdx.x * blockDim.x + threadIdx.x;
  if (idx >= Bsz * Hn) return;
  int b = idx >> 9, n = idx & (Hn - 1);
  size_t o = ((size_t)b * SLOTS + ZSLOT) * Hn + n;
  cf[o] = 0.0f; hf[o] = 0.0f; hb[o] = 0;
}

// ---------------- leaf phase: fused dual WMMA GEMM + activations ----------------

__global__ void k_leaf(const unsigned short* __restrict__ xb,
                       const unsigned short* __restrict__ Wpt,
                       const unsigned short* __restrict__ Wgt,
                       const float* __restrict__ bproj, const float* __restrict__ bgate,
                       float* __restrict__ cf, float* __restrict__ hf,
                       unsigned short* __restrict__ hb) {
  int wave = threadIdx.x >> 5, lane = threadIdx.x & 31;
  int hi = lane >> 4, ln = lane & 15;
  int m0 = blockIdx.x * 32 + (wave >> 2) * 16;   // gridDim.x = MT/32
  int n0 = blockIdx.y * 64 + (wave & 3) * 16;    // gridDim.y = Hn/64

  const unsigned short* arow = xb  + (size_t)(m0 + ln) * KP;
  const unsigned short* prow = Wpt + (size_t)(n0 + ln) * KP;
  const unsigned short* grow = Wgt + (size_t)(n0 + ln) * KP;

  v8f accP = {}; v8f accG = {};
  for (int kb = 0; kb < KP; kb += 32) {
    Frag a, bP, bG;
#pragma unroll
    for (int p = 0; p < 8; ++p) {
      int o = kb + KOFF(p, hi);
      a.w[p]  = *(const unsigned int*)(arow + o);
      bP.w[p] = *(const unsigned int*)(prow + o);
      bG.w[p] = *(const unsigned int*)(grow + o);
    }
    accP = __builtin_amdgcn_wmma_f32_16x16x32_bf16(false, a.v, false, bP.v, (short)0, accP, false, false);
    accG = __builtin_amdgcn_wmma_f32_16x16x32_bf16(false, a.v, false, bG.v, (short)0, accG, false, false);
  }
#pragma unroll
  for (int r = 0; r < 8; ++r) {
    int m = m0 + r + hi * 8, col = n0 + ln;
    int b = m >> 7, tt = m & (Tn - 1);           // leaf slot tt for batch row b
    float c = accP[r] + bproj[col];
    float g = accG[r] + bgate[col];
    float h = sigmoidf_(g) * tanhf(c);
    size_t o = ((size_t)b * SLOTS + tt) * Hn + col;
    cf[o] = c; hf[o] = h; hb[o] = f2bf(h);
  }
}

// ---------------- scan phase: one fully fused kernel per step ----------------
// 5-gate GEMM (M=256,K=1024,N=512 per gate) + bias + cell nonlinearity + slot write.
// Wave tile: 32 rows x 16 cols, all 5 gates. Block: 8 waves (2 mi x 4 nj) = 64x64.
__global__ void k_reduce_step(const unsigned short* __restrict__ hb,
                              const float* __restrict__ cf,
                              float* __restrict__ cfo, float* __restrict__ hfo,
                              unsigned short* __restrict__ hbo,
                              const unsigned short* __restrict__ Wrt,
                              const float* __restrict__ br,
                              const int* __restrict__ lslot,
                              const int* __restrict__ rslot,
                              const int* __restrict__ flags, int t) {
  if (flags[t] == 0) return;                     // all-shift step: nothing to do
  int wave = threadIdx.x >> 5, lane = threadIdx.x & 31;
  int hi = lane >> 4, ln = lane & 15;
  int mi = wave >> 2, nj = wave & 3;
  int m0 = blockIdx.x * 64 + mi * 32;            // gridDim.x = Bsz/64 = 4
  int n0 = blockIdx.y * 64 + nj * 16;            // gridDim.y = Hn/64  = 8
  int col = n0 + ln;

  // A row pointers (gathered via precomputed slots) for 2 row-subtiles
  const unsigned short* pL[2]; const unsigned short* pR[2];
#pragma unroll
  for (int s = 0; s < 2; ++s) {
    int row = m0 + s * 16 + ln;
    int ls = lslot[t * Bsz + row], rs = rslot[t * Bsz + row];
    pL[s] = hb + ((size_t)row * SLOTS + ls) * Hn;
    pR[s] = hb + ((size_t)row * SLOTS + rs) * Hn;
  }
  // B pointers: gate j, column col of W_reduce (transposed layout, K contiguous)
  const unsigned short* pB[5];
#pragma unroll
  for (int j = 0; j < 5; ++j) pB[j] = Wrt + ((size_t)(j * Hn + col)) * KR;

  v8f acc[2][5];
#pragma unroll
  for (int j = 0; j < 5; ++j) {
    float bias = br[j * Hn + col];
#pragma unroll
    for (int s = 0; s < 2; ++s)
#pragma unroll
      for (int r = 0; r < 8; ++r) acc[s][j][r] = bias;
  }

  for (int kb = 0; kb < KR; kb += 32) {
    const unsigned short* pa0 = (kb < Hn) ? (pL[0] + kb) : (pR[0] + kb - Hn);
    const unsigned short* pa1 = (kb < Hn) ? (pL[1] + kb) : (pR[1] + kb - Hn);
    Frag a0, a1, bf[5];
#pragma unroll
    for (int p = 0; p < 8; ++p) {
      int o = KOFF(p, hi);
      a0.w[p] = *(const unsigned int*)(pa0 + o);
      a1.w[p] = *(const unsigned int*)(pa1 + o);
    }
#pragma unroll
    for (int j = 0; j < 5; ++j)
#pragma unroll
      for (int p = 0; p < 8; ++p)
        bf[j].w[p] = *(const unsigned int*)(pB[j] + kb + KOFF(p, hi));
#pragma unroll
    for (int j = 0; j < 5; ++j) {
      acc[0][j] = __builtin_amdgcn_wmma_f32_16x16x32_bf16(false, a0.v, false, bf[j].v, (short)0, acc[0][j], false, false);
      acc[1][j] = __builtin_amdgcn_wmma_f32_16x16x32_bf16(false, a1.v, false, bf[j].v, (short)0, acc[1][j], false, false);
    }
  }

  // epilogue: cell nonlinearity, write fresh slot Tn+t (no hazards: all reads
  // are from strictly older slots, so full fusion is race-free)
  int w = Tn + t;
#pragma unroll
  for (int s = 0; s < 2; ++s) {
#pragma unroll
    for (int r = 0; r < 8; ++r) {
      int row = m0 + s * 16 + r + hi * 8;
      int ls = lslot[t * Bsz + row], rs = rslot[t * Bsz + row];
      float cl = cf[((size_t)row * SLOTS + ls) * Hn + col];
      float cr = cf[((size_t)row * SLOTS + rs) * Hn + col];
      float i_ = acc[s][0][r], fl = acc[s][1][r], fr = acc[s][2][r];
      float g  = acc[s][3][r], o_ = acc[s][4][r];
      float c = sigmoidf_(fl) * cl + sigmoidf_(fr) * cr + sigmoidf_(i_) * tanhf(g);
      float h = sigmoidf_(o_) * tanhf(c);
      size_t oo = ((size_t)row * SLOTS + w) * Hn + col;
      cfo[oo] = c; hfo[oo] = h; hbo[oo] = f2bf(h);
    }
  }
}

__global__ void k_out(const float* __restrict__ hf, const int* __restrict__ top,
                      float* __restrict__ out) {
  int idx = blockIdx.x * blockDim.x + threadIdx.x;
  if (idx >= Bsz * Hn) return;
  int b = idx >> 9, n = idx & (Hn - 1);
  out[idx] = hf[((size_t)b * SLOTS + top[b]) * Hn + n];
}

// ---------------- host ----------------

static inline size_t align_up256(size_t v) { return (v + 255) & ~(size_t)255; }

extern "C" void kernel_launch(void* const* d_in, const int* in_sizes, int n_in,
                              void* d_out, int out_size, void* d_ws, size_t ws_size,
                              hipStream_t stream) {
  (void)n_in; (void)out_size; (void)ws_size;
  const float* x     = (const float*)d_in[0];
  const float* Wp    = (const float*)d_in[1];
  const float* bpj   = (const float*)d_in[2];
  const float* Wg    = (const float*)d_in[3];
  const float* bg    = (const float*)d_in[4];
  const float* Wr    = (const float*)d_in[5];
  const float* br    = (const float*)d_in[6];
  const int*   trans = (const int*)d_in[7];
  float* out = (float*)d_out;

  int steps = in_sizes[7] / Bsz;                 // 2T-1 = 255
  if (steps > SMAX - 1) steps = SMAX - 1;

  char* w = (char*)d_ws; size_t off = 0;
  auto alloc = [&](size_t bytes) -> void* {
    void* p = w + off; off = align_up256(off + bytes); return p;
  };
  unsigned short* xb    = (unsigned short*)alloc((size_t)MT * KP * 2);
  unsigned short* Wpt   = (unsigned short*)alloc((size_t)Hn * KP * 2);
  unsigned short* Wgt   = (unsigned short*)alloc((size_t)Hn * KP * 2);
  unsigned short* Wrt   = (unsigned short*)alloc((size_t)NR * KR * 2);
  float*          cfA   = (float*)alloc((size_t)Bsz * SLOTS * Hn * 4);          // c arena (f32)
  float*          hfA   = (float*)alloc((size_t)Bsz * SLOTS * Hn * 4);          // h arena (f32)
  unsigned short* hbA   = (unsigned short*)alloc((size_t)Bsz * SLOTS * Hn * 2); // h arena (bf16)
  int*            lslot = (int*)alloc((size_t)SMAX * Bsz * 4);
  int*            rslot = (int*)alloc((size_t)SMAX * Bsz * 4);
  int*            top   = (int*)alloc(256 * 4);
  int*            flg   = (int*)alloc(256 * 4);

  // schedule + conversions + zero slot
  k_sched<<<1, 256, 0, stream>>>(trans, steps, lslot, rslot, top, flg);
  k_conv_x<<<(int)(((size_t)MT * KP + 255) / 256), 256, 0, stream>>>(x, xb);
  k_conv_wpg<<<(int)(((size_t)Hn * KP + 255) / 256), 256, 0, stream>>>(Wp, Wg, Wpt, Wgt);
  k_conv_wr<<<(int)(((size_t)NR * KR + 255) / 256), 256, 0, stream>>>(Wr, Wrt);
  k_zero_z<<<(Bsz * Hn + 255) / 256, 256, 0, stream>>>(cfA, hfA, hbA);

  // leaf GEMMs + activations (WMMA bf16) -> leaf slots 0..Tn-1
  {
    dim3 grid(MT / 32, Hn / 64);
    k_leaf<<<grid, 256, 0, stream>>>(xb, Wpt, Wgt, bpj, bg, cfA, hfA, hbA);
  }

  // sequential shift/reduce scan: one fused kernel per step
  for (int t = 0; t < steps; ++t) {
    dim3 grid(Bsz / 64, Hn / 64);
    k_reduce_step<<<grid, 256, 0, stream>>>(hbA, cfA, cfA, hfA, hbA, Wrt, br,
                                            lslot, rslot, flg, t);
  }

  k_out<<<(Bsz * Hn + 255) / 256, 256, 0, stream>>>(hfA, top, out);
}
